// Model_85160611545155
// MI455X (gfx1250) — matrix operand order repaired and checked by reading, not compiled, not run on hardware
//
#include <hip/hip_runtime.h>
#include <hip/hip_bf16.h>

typedef __bf16 bf16_t;
typedef __attribute__((ext_vector_type(16))) __bf16 v16bf;
typedef __attribute__((ext_vector_type(8)))  __bf16 v8bf;
typedef __attribute__((ext_vector_type(8)))  float  v8f;

#define B_   128
#define S_   512
#define C_   64
#define H_   512
#define P_   192
#define G4H_ (4 * H_)
#define SP_  (S_ + P_)        // 704
#define SPM1_ (S_ + P_ - 1)   // 703

// ---------------------------------------------------------------------------
// helpers
// ---------------------------------------------------------------------------
static __device__ __forceinline__ v16bf cat8(v8bf lo, v8bf hi) {
  return __builtin_shufflevector(lo, hi, 0, 1, 2, 3, 4, 5, 6, 7,
                                         8, 9, 10, 11, 12, 13, 14, 15);
}

static __device__ __forceinline__ v8f wmma_bf16(v16bf a, v16bf b, v8f c) {
  // D = A(16x32 bf16) * B(32x16 bf16) + C(16x16 f32)
  return __builtin_amdgcn_wmma_f32_16x16x32_bf16(
      false, a, false, b, (short)0, c, false, false);
}

// 16x32 bf16 A-tile slice: lanes 0-15 hold K 0-15 of row (lane&15),
// lanes 16-31 hold K 16-31.
static __device__ __forceinline__ v16bf load_a(const bf16_t* rowptr, int hi) {
  return *(const v16bf*)(rowptr + hi * 16);
}

// 32x16 bf16 B-tile slice from row-major W[N,K]: lane owns column n,
// lanes 0-15 K 0-15, lanes 16-31 K 16-31 -> one contiguous 32B run.
static __device__ __forceinline__ v16bf load_b(const bf16_t* wrow, int hi) {
  return *(const v16bf*)(wrow + hi * 16);
}

static __device__ __forceinline__ float sigm(float x) {
  return 1.0f / (1.0f + __expf(-x));
}

// ---------------------------------------------------------------------------
// dtype conversion / bias prep
// ---------------------------------------------------------------------------
__global__ void f32_to_bf16_kernel(const float* __restrict__ in,
                                   bf16_t* __restrict__ out, int n) {
  int i = blockIdx.x * blockDim.x + threadIdx.x;
  if (i < n) out[i] = (bf16_t)in[i];
}

__global__ void bias_sum_kernel(const float* __restrict__ a,
                                const float* __restrict__ b,
                                float* __restrict__ o, int n) {
  int i = blockIdx.x * blockDim.x + threadIdx.x;
  if (i < n) o[i] = a[i] + b[i];
}

// ---------------------------------------------------------------------------
// Encoder: XE[t,b,:] = x[b,t,:] @ encW^T + enc_b   (stored [S,B,H] bf16)
// ---------------------------------------------------------------------------
__global__ void encoder_kernel(const bf16_t* __restrict__ Xbf,   // [B,S,C]
                               const bf16_t* __restrict__ Wenc,  // [H,C]
                               const float* __restrict__ benc,   // [H]
                               bf16_t* __restrict__ XE) {        // [S,B,H]
  const int tid = threadIdx.x, lane = tid & 31;
  const int wid = blockIdx.x * (blockDim.x >> 5) + (tid >> 5);
  const int m_tile = wid >> 5;   // M = S*B rows -> 4096 tiles
  const int n_tile = wid & 31;   // H/16 = 32 tiles
  const int hi = lane >> 4, lr = lane & 15;

  const int m = m_tile * 16 + lr;
  const int t = m / B_, b = m % B_;
  const int ncol = n_tile * 16 + lr;

  const bf16_t* arow = Xbf + ((size_t)b * S_ + t) * C_;
  const bf16_t* wrow = Wenc + (size_t)ncol * C_;

  v8f acc = {};
#pragma unroll
  for (int k0 = 0; k0 < C_; k0 += 32) {
    acc = wmma_bf16(load_a(arow + k0, hi), load_b(wrow + k0, hi), acc);
  }

  const float bv = benc[ncol];
#pragma unroll
  for (int r = 0; r < 8; ++r) {
    int mm = m_tile * 16 + r + hi * 8;
    int tt = mm / B_, bb = mm % B_;
    XE[((size_t)tt * B_ + bb) * H_ + ncol] = (bf16_t)(acc[r] + bv);
  }
}

// ---------------------------------------------------------------------------
// Bulk input GEMM (parallel, off the critical path):
//   XW[t,b,:] = XE[t,b,:] @ Wih^T   (fp32, [S,B,4H])
// one wave = one 16x16 tile; rows of XE[S,B,H] are contiguous in m = t*B+b.
// ---------------------------------------------------------------------------
__global__ void xw_gemm_kernel(const bf16_t* __restrict__ XE,   // [S*B, H]
                               const bf16_t* __restrict__ Wih,  // [4H, H]
                               float* __restrict__ XW) {        // [S*B, 4H]
  const int tid = threadIdx.x, lane = tid & 31;
  const int wid = blockIdx.x * (blockDim.x >> 5) + (tid >> 5);
  const int m_tile = wid >> 7;    // 4096 M tiles
  const int n_tile = wid & 127;   // 4H/16 = 128 N tiles
  const int hi = lane >> 4, lr = lane & 15;

  const int ncol = n_tile * 16 + lr;                 // in [0, 4H)
  const bf16_t* arow = XE + (size_t)(m_tile * 16 + lr) * H_;
  const bf16_t* wrow = Wih + (size_t)ncol * H_;

  v8f acc = {};
#pragma unroll 4
  for (int k0 = 0; k0 < H_; k0 += 32) {
    acc = wmma_bf16(load_a(arow + k0, hi), load_b(wrow + k0, hi), acc);
  }

#pragma unroll
  for (int r = 0; r < 8; ++r) {
    int mm = m_tile * 16 + r + hi * 8;
    XW[(size_t)mm * G4H_ + ncol] = acc[r];
  }
}

// ---------------------------------------------------------------------------
// Unified LSTM cell step.
//   pre = (usePre ? Pre : 0) + (useS ? As @ Ws^T : 0) + (useG ? Ag @ Wg^T : 0)
//         + bias;  then i,f,g,o gates; c' = f*c + i*g; h = o*tanh(c')
// Grid: 32 blocks (one per 16-col h tile) x 256 thr (8 waves = 8 M tiles).
// Ws columns for all 4 gates are staged in exactly 64KB of LDS (shared by
// the 8 waves), with a per-row rotation swizzle to avoid bank conflicts.
// ---------------------------------------------------------------------------
__global__ void lstm_cell_kernel(const float* __restrict__ Pre,  // [B,4H] f32
                                 const bf16_t* __restrict__ As,  // [B,H]
                                 const bf16_t* __restrict__ Ws,  // [4H,H]
                                 const bf16_t* __restrict__ Ag,  // [B,H]
                                 const bf16_t* __restrict__ Wg,  // [4H,H]
                                 const float* __restrict__ bias, // [4H]
                                 float* __restrict__ Cst,        // [B,H] f32
                                 bf16_t* __restrict__ Hout,      // [B,H] bf16
                                 int usePre, int useS, int useG,
                                 int haveCin, int writeC) {
  extern __shared__ char smem[];
  bf16_t* lds = (bf16_t*)smem;     // [64 rows][512 halves], rotated chunks

  const int tid = threadIdx.x, lane = tid & 31;
  const int m_tile = tid >> 5;     // 0..7
  const int n_tile = blockIdx.x;   // 0..31
  const int hi = lane >> 4, lr = lane & 15;
  const int m0 = m_tile * 16;
  const int n0 = n_tile * 16;
  const int ncol = n0 + lr;

  // ---- cooperative staging of Ws columns (4 gates x 16 cols x 512 K) ----
  if (useS) {
#pragma unroll
    for (int it = 0; it < 16; ++it) {            // 4096 chunks / 256 threads
      int idx = it * 256 + tid;
      int row = idx >> 6;                        // g*16 + col
      int ck  = idx & 63;                        // 8-half chunk within K
      v8bf v = *(const v8bf*)(Ws +
          (size_t)((row >> 4) * H_ + n0 + (row & 15)) * H_ + ck * 8);
      int ckr = (ck + row * 2) & 63;             // rotation swizzle
      *(v8bf*)(lds + row * H_ + ckr * 8) = v;
    }
  }
  __syncthreads();

  // ---- accumulators (optionally initialized from precomputed XW) ----
  v8f acc[4] = {v8f{}, v8f{}, v8f{}, v8f{}};
  if (usePre) {
#pragma unroll
    for (int g = 0; g < 4; ++g)
#pragma unroll
      for (int r = 0; r < 8; ++r)
        acc[g][r] = Pre[(size_t)(m0 + r + hi * 8) * G4H_ + g * H_ + ncol];
  }

  const bf16_t* asrow = As + (size_t)(m0 + lr) * H_;
  const bf16_t* agrow = Ag + (size_t)(m0 + lr) * H_;
  const bf16_t* wg0 = Wg + (size_t)(0 * H_ + ncol) * H_;
  const bf16_t* wg1 = Wg + (size_t)(1 * H_ + ncol) * H_;
  const bf16_t* wg2 = Wg + (size_t)(2 * H_ + ncol) * H_;
  const bf16_t* wg3 = Wg + (size_t)(3 * H_ + ncol) * H_;

#pragma unroll 4
  for (int k0 = 0; k0 < H_; k0 += 32) {
    if (useS) {
      v16bf a = load_a(asrow + k0, hi);
#pragma unroll
      for (int g = 0; g < 4; ++g) {
        int row = g * 16 + lr;
        int ch = (((k0 >> 3) + (hi << 1)) + row * 2) & 63;
        v16bf b = *(const v16bf*)(lds + row * H_ + ch * 8);
        acc[g] = wmma_bf16(a, b, acc[g]);
      }
    }
    if (useG) {
      v16bf a = load_a(agrow + k0, hi);
      acc[0] = wmma_bf16(a, load_b(wg0 + k0, hi), acc[0]);
      acc[1] = wmma_bf16(a, load_b(wg1 + k0, hi), acc[1]);
      acc[2] = wmma_bf16(a, load_b(wg2 + k0, hi), acc[2]);
      acc[3] = wmma_bf16(a, load_b(wg3 + k0, hi), acc[3]);
      if (k0 + 32 < H_) {
        __builtin_prefetch((const void*)(wg0 + k0 + 32), 0, 1);
        __builtin_prefetch((const void*)(wg2 + k0 + 32), 0, 1);
      }
    }
  }

  const float b0 = bias[0 * H_ + ncol];
  const float b1 = bias[1 * H_ + ncol];
  const float b2 = bias[2 * H_ + ncol];
  const float b3 = bias[3 * H_ + ncol];

#pragma unroll
  for (int r = 0; r < 8; ++r) {
    const int brow = m0 + r + hi * 8;           // D-layout row = batch index
    const size_t idx = (size_t)brow * H_ + ncol;
    float iv = sigm(acc[0][r] + b0);
    float fv = sigm(acc[1][r] + b1);
    float gv = tanhf(acc[2][r] + b2);
    float ov = sigm(acc[3][r] + b3);
    float cold = haveCin ? Cst[idx] : 0.0f;
    float cnew = fv * cold + iv * gv;
    if (writeC) Cst[idx] = cnew;
    Hout[idx] = (bf16_t)(ov * tanhf(cnew));
  }
}

// ---------------------------------------------------------------------------
// Projection: out[b,j,:] = Hbuf[1+j, b, :] @ decW^T + dec_b, j in [0,703)
// ---------------------------------------------------------------------------
__global__ void proj_kernel(const bf16_t* __restrict__ Hbuf,  // [S+P,B,H]
                            const bf16_t* __restrict__ Wdec,  // [C,H]
                            const float* __restrict__ bdec,   // [C]
                            float* __restrict__ Out) {        // [B,703,C]
  const int tid = threadIdx.x, lane = tid & 31;
  const int wid = blockIdx.x * (blockDim.x >> 5) + (tid >> 5);
  const int m_tile = wid >> 2;   // M = B*703 rows -> 5624 tiles
  const int n_tile = wid & 3;    // C/16 = 4 tiles
  const int hi = lane >> 4, lr = lane & 15;

  const int m = m_tile * 16 + lr;
  const int b = m / SPM1_, j = m % SPM1_;
  const int ncol = n_tile * 16 + lr;

  const bf16_t* arow = Hbuf + ((size_t)(j + 1) * B_ + b) * H_;
  const bf16_t* wrow = Wdec + (size_t)ncol * H_;

  v8f acc = {};
#pragma unroll 4
  for (int k0 = 0; k0 < H_; k0 += 32) {
    acc = wmma_bf16(load_a(arow + k0, hi), load_b(wrow + k0, hi), acc);
  }

  const float bv = bdec[ncol];
#pragma unroll
  for (int r = 0; r < 8; ++r) {
    int mm = m_tile * 16 + r + hi * 8;
    int bb = mm / SPM1_, jj = mm % SPM1_;
    Out[((size_t)bb * SPM1_ + jj) * C_ + ncol] = acc[r] + bv;
  }
}

// ---------------------------------------------------------------------------
// launch
// ---------------------------------------------------------------------------
extern "C" void kernel_launch(void* const* d_in, const int* in_sizes, int n_in,
                              void* d_out, int out_size, void* d_ws,
                              size_t ws_size, hipStream_t stream) {
  (void)in_sizes; (void)n_in; (void)out_size;

  const float* x    = (const float*)d_in[0];  // [B,S,C]
  const float* encW = (const float*)d_in[1];  // [H,C]
  const float* encb = (const float*)d_in[2];  // [H]
  const float* Wih  = (const float*)d_in[3];  // [4H,H]
  const float* Whh  = (const float*)d_in[4];  // [4H,H]
  const float* bih  = (const float*)d_in[5];  // [4H]
  const float* bhh  = (const float*)d_in[6];  // [4H]
  const float* decW = (const float*)d_in[7];  // [C,H]
  const float* decb = (const float*)d_in[8];  // [C]
  float* out = (float*)d_out;

  // workspace carve-out (256B aligned chunks)
  char* ws = (char*)d_ws;
  size_t off = 0;
  auto carve = [&](size_t bytes) -> void* {
    void* p = ws + off;
    off += (bytes + 255) & ~(size_t)255;
    return p;
  };
  bf16_t* xbf   = (bf16_t*)carve((size_t)B_ * S_ * C_ * 2);
  bf16_t* XE    = (bf16_t*)carve((size_t)S_ * B_ * H_ * 2);
  bf16_t* Hbuf  = (bf16_t*)carve((size_t)SP_ * B_ * H_ * 2);
  bf16_t* WihB  = (bf16_t*)carve((size_t)G4H_ * H_ * 2);
  bf16_t* WhhB  = (bf16_t*)carve((size_t)G4H_ * H_ * 2);
  bf16_t* encWB = (bf16_t*)carve((size_t)H_ * C_ * 2);
  bf16_t* decWB = (bf16_t*)carve((size_t)C_ * H_ * 2);
  float*  Cst   = (float*)carve((size_t)B_ * H_ * 4);
  float*  bias  = (float*)carve((size_t)G4H_ * 4);

  // Fast path: hoist xe@Wih^T off the serial chain into one parallel GEMM.
  const size_t xw_bytes = (size_t)S_ * B_ * G4H_ * 4;  // 537 MB fp32
  const bool fast = (ws_size >= off + xw_bytes + 256);
  float* XW = fast ? (float*)carve(xw_bytes) : nullptr;

  // ---- precision conversion (weights stay hot in L2 as bf16) ----
  auto cvt = [&](const float* src, bf16_t* dst, int n) {
    f32_to_bf16_kernel<<<(n + 255) / 256, 256, 0, stream>>>(src, dst, n);
  };
  cvt(x,    xbf,   B_ * S_ * C_);
  cvt(encW, encWB, H_ * C_);
  cvt(Wih,  WihB,  G4H_ * H_);
  cvt(Whh,  WhhB,  G4H_ * H_);
  cvt(decW, decWB, C_ * H_);
  bias_sum_kernel<<<(G4H_ + 255) / 256, 256, 0, stream>>>(bih, bhh, bias, G4H_);

  // ---- encoder GEMM ----
  encoder_kernel<<<16384, 256, 0, stream>>>(xbf, encWB, encb, XE);

  const size_t BH = (size_t)B_ * H_;
  const int ldsW = 4 * 16 * H_ * 2;   // 64KB staged weight region

  if (fast) {
    // big parallel GEMM: XW = XE @ Wih^T  (524288 tile-waves)
    xw_gemm_kernel<<<65536, 256, 0, stream>>>(XE, WihB, XW);

    // serial LSTM: each step only h @ Whh^T (+ XW pre-accumulator)
    for (int t = 0; t < S_; ++t) {
      lstm_cell_kernel<<<32, 256, t ? ldsW : 0, stream>>>(
          XW + (size_t)t * B_ * G4H_,
          t ? (Hbuf + (size_t)(t - 1) * BH) : XE, WhhB,   // staged stream
          XE, WhhB,                                       // unused global stream
          bias, Cst, Hbuf + (size_t)t * BH,
          /*usePre=*/1, /*useS=*/t ? 1 : 0, /*useG=*/0,
          /*haveCin=*/t ? 1 : 0, /*writeC=*/1);
    }
  } else {
    // fused fallback: both GEMMs inside each step (Wih staged, Whh global)
    for (int t = 0; t < S_; ++t) {
      lstm_cell_kernel<<<32, 256, ldsW, stream>>>(
          Cst /*unused*/,
          XE + (size_t)t * BH, WihB,                      // staged stream
          t ? (Hbuf + (size_t)(t - 1) * BH) : XE, WhhB,   // global stream
          bias, Cst, Hbuf + (size_t)t * BH,
          /*usePre=*/0, /*useS=*/1, /*useG=*/t ? 1 : 0,
          /*haveCin=*/t ? 1 : 0, /*writeC=*/1);
    }
  }

  // ---- autoregressive decode: zero-state cell fed previous h ----
  for (int p = 0; p < P_; ++p) {
    lstm_cell_kernel<<<32, 256, ldsW, stream>>>(
        Cst /*unused*/,
        Hbuf + (size_t)(S_ - 1 + p) * BH, WihB,           // staged stream
        XE, WhhB,                                         // unused global stream
        bias, Cst, Hbuf + (size_t)(S_ + p) * BH,
        /*usePre=*/0, /*useS=*/1, /*useG=*/0,
        /*haveCin=*/0, /*writeC=*/0);
  }

  // ---- output projection over the 703 kept timesteps ----
  proj_kernel<<<2812, 256, 0, stream>>>(Hbuf, decWB, decb, out);
}
